// RoadNetworkEncoder_44083544326738
// MI455X (gfx1250) — matrix operand order repaired and checked by reading, not compiled, not run on hardware
//
#include <hip/hip_runtime.h>
#include <hip/hip_bf16.h>

typedef __attribute__((ext_vector_type(16))) _Float16 v16h;
typedef __attribute__((ext_vector_type(8)))  _Float16 v8h;
typedef __attribute__((ext_vector_type(4)))  _Float16 v4h;
typedef __attribute__((ext_vector_type(8)))  float    v8f;

constexpr int N_ROAD = 100000, E_ROAD = 400000;
constexpr int N_ZONE = 20000,  E_ZONE = 160000;
constexpr int H = 128;
constexpr int E2R = E_ROAD + N_ROAD;   // road edges + self loops
constexpr int E2Z = E_ZONE + N_ZONE;   // zone edges + self loops

static inline int cdiv(long long a, long long b) { return (int)((a + b - 1) / b); }

// ---------------------------------------------------------------------------
// Road node feature assembly: [id_emb(64) | len lin(16) | type_emb(16) |
//                              lon lin(16) | lat lin(16)] -> f16 activations
// ---------------------------------------------------------------------------
__global__ void k_build_feats(const float* __restrict__ id_emb,
                              const float* __restrict__ lenW, const float* __restrict__ lenb,
                              const float* __restrict__ type_emb,
                              const float* __restrict__ lonW, const float* __restrict__ lonb,
                              const float* __restrict__ latW, const float* __restrict__ latb,
                              const float* __restrict__ alen, const int* __restrict__ atype,
                              const float* __restrict__ alon, const float* __restrict__ alat,
                              _Float16* __restrict__ xh)
{
    long long tid = (long long)blockIdx.x * blockDim.x + threadIdx.x;
    if (tid >= (long long)N_ROAD * H) return;
    int node = (int)(tid >> 7), j = (int)(tid & 127);
    float v;
    if (j < 64)       v = id_emb[(size_t)node * 64 + j];
    else if (j < 80)  v = alen[node] * lenW[j - 64] + lenb[j - 64];
    else if (j < 96)  v = type_emb[(size_t)atype[node] * 16 + (j - 80)];
    else if (j < 112) v = alon[node] * lonW[j - 96] + lonb[j - 96];
    else              v = alat[node] * latW[j - 112] + latb[j - 112];
    xh[tid] = (_Float16)v;
}

// W [128(k) x 128(n)] f32  ->  Wt [n][k] f16 (so B fragments read contiguously)
__global__ void k_wt(const float* __restrict__ W, _Float16* __restrict__ Wt)
{
    int tid = blockIdx.x * blockDim.x + threadIdx.x; // 16384
    if (tid >= H * H) return;
    int n = tid >> 7, k = tid & 127;
    Wt[n * H + k] = (_Float16)W[k * H + n];
}

// f32 -> f16, 4 elements per thread (b128 load, 8B store)
__global__ void k_f2h4(const float* __restrict__ x, _Float16* __restrict__ xh, long long n4)
{
    long long t = (long long)blockIdx.x * blockDim.x + threadIdx.x;
    if (t >= n4) return;
    float4 v = ((const float4*)x)[t];
    v4h o;
    o[0] = (_Float16)v.x; o[1] = (_Float16)v.y;
    o[2] = (_Float16)v.z; o[3] = (_Float16)v.w;
    ((v4h*)xh)[t] = o;
}

// ---------------------------------------------------------------------------
// WMMA GEMM: Y[N,128] = A[N,128](f16) @ W(via Wt f16) [+ bias]   (f32 accum)
// 256 threads = 8 waves; wave w -> output cols [16w,16w+16). B fragments held
// in registers across a grid-stride loop over 16-row A tiles staged in LDS.
// ---------------------------------------------------------------------------
__global__ __launch_bounds__(256) void k_gemm128(const _Float16* __restrict__ A,
                                                 const _Float16* __restrict__ Bt,
                                                 const float* __restrict__ bias,
                                                 float* __restrict__ Y, int nTiles)
{
    __shared__ __align__(16) _Float16 lds_a[16 * H];
    const int wave = threadIdx.x >> 5;
    const int lane = threadIdx.x & 31;
    const int n    = lane & 15;      // column within 16x16 tile (== A row m)
    const int g    = lane >> 4;      // lane-group (half-wave)
    const int col  = wave * 16 + n;

    // B k order used here (contiguous, not AMD's table order): lanes 0-15 K 0..15, lanes 16-31 K 16..31,
    // ascending within the 8 VGPRs -> element e <-> k = ks*32 + g*16 + e.
    v16h bf[4];
#pragma unroll
    for (int ks = 0; ks < 4; ++ks)
        bf[ks] = *(const v16h*)(Bt + (size_t)col * H + ks * 32 + g * 16);

    const float bcol = bias ? bias[col] : 0.0f;

    for (int rt = blockIdx.x; rt < nTiles; rt += gridDim.x) {
        __syncthreads();
        *(v8h*)(lds_a + threadIdx.x * 8) =
            *(const v8h*)(A + (size_t)rt * 16 * H + threadIdx.x * 8);
        __syncthreads();

        v8f acc = {0.f, 0.f, 0.f, 0.f, 0.f, 0.f, 0.f, 0.f};
#pragma unroll
        for (int ks = 0; ks < 4; ++ks) {
            // A k order used here (contiguous, matches B): e in [0,16) <-> k = k0 + g*16 + e
            v8h lo = *(const v8h*)(lds_a + n * H + ks * 32 + g * 16);
            v8h hi = *(const v8h*)(lds_a + n * H + ks * 32 + g * 16 + 8);
            v16h a;
#pragma unroll
            for (int i = 0; i < 8; ++i) { a[i] = lo[i]; a[i + 8] = hi[i]; }
            acc = __builtin_amdgcn_wmma_f32_16x16x32_f16(
                false, a, false, bf[ks], (short)0, acc, false, false);
        }
        // C/D layout: VGPR r <-> row (rt*16 + 8g + r), lane%16 <-> col.
        // One base pointer; r*H folds into the store's immediate offset.
        float* yp = Y + ((size_t)(rt * 16 + 8 * g)) * H + col;
#pragma unroll
        for (int r = 0; r < 8; ++r)
            yp[r * H] = acc[r] + bcol;
    }
}

// ---------------------------------------------------------------------------
// GATv2 support kernels
// ---------------------------------------------------------------------------
__global__ void k_ea_accum(const int* __restrict__ ei, const float* __restrict__ attr,
                           float* __restrict__ easum, float* __restrict__ cnt)
{
    int e = blockIdx.x * blockDim.x + threadIdx.x;
    if (e >= E_ROAD) return;
    int dst = ei[E_ROAD + e];
    atomicAdd(&cnt[dst], 1.0f);
    atomicAdd(&easum[dst * 2 + 0], attr[2 * e + 0]);
    atomicAdd(&easum[dst * 2 + 1], attr[2 * e + 1]);
}

__global__ void k_ea_div(float* __restrict__ easum, const float* __restrict__ cnt)
{
    int i = blockIdx.x * blockDim.x + threadIdx.x;
    if (i >= N_ROAD) return;
    float c = fmaxf(cnt[i], 1.0f);
    easum[i * 2 + 0] /= c;
    easum[i * 2 + 1] /= c;
}

// one wave per edge: logit = dot(leaky_relu(xl[src]+xr[dst]+ea@We), att)
__global__ __launch_bounds__(256) void k_gat_logits(
    const float* __restrict__ xl, const float* __restrict__ xr,
    const int* __restrict__ ei, const float* __restrict__ attr,
    const float* __restrict__ mean_ea,
    const float* __restrict__ We, const float* __restrict__ att,
    float* __restrict__ elog, unsigned int* __restrict__ maxkey)
{
    int e    = blockIdx.x * 8 + (threadIdx.x >> 5);
    int lane = threadIdx.x & 31;
    if (e >= E2R) return;
    int src, dst; float ea0, ea1;
    if (e < E_ROAD) {
        src = ei[e]; dst = ei[E_ROAD + e];
        ea0 = attr[2 * e]; ea1 = attr[2 * e + 1];
    } else {
        src = dst = e - E_ROAD;
        ea0 = mean_ea[2 * src]; ea1 = mean_ea[2 * src + 1];
    }
    const float* pl = xl + (size_t)src * H;
    const float* pr = xr + (size_t)dst * H;
    float s = 0.f;
#pragma unroll
    for (int t = 0; t < 4; ++t) {
        int j = lane + 32 * t;
        float v = pl[j] + pr[j] + ea0 * We[j] + ea1 * We[H + j];
        v = v > 0.f ? v : 0.2f * v;              // leaky_relu(0.2)
        s += v * att[j];
    }
#pragma unroll
    for (int off = 16; off; off >>= 1) s += __shfl_down(s, off, 32);
    if (lane == 0) {
        elog[e] = s;
        unsigned int b = __float_as_uint(s);
        unsigned int key = (b & 0x80000000u) ? ~b : (b ^ 0x80000000u); // order-preserving
        atomicMax(&maxkey[dst], key);
    }
}

__global__ void k_gat_exp(const int* __restrict__ ei, float* __restrict__ elog,
                          const unsigned int* __restrict__ maxkey, float* __restrict__ denom)
{
    int e = blockIdx.x * blockDim.x + threadIdx.x;
    if (e >= E2R) return;
    int dst = (e < E_ROAD) ? ei[E_ROAD + e] : (e - E_ROAD);
    unsigned int key = maxkey[dst];
    unsigned int b = (key & 0x80000000u) ? (key ^ 0x80000000u) : ~key;
    float ex = __expf(elog[e] - __uint_as_float(b));
    elog[e] = ex;
    atomicAdd(&denom[dst], ex);
}

// 32 threads per edge, float4 per thread: b128 gather + 4 f32 atomics
__global__ void k_gat_aggr(const int* __restrict__ ei, const float* __restrict__ elog,
                           const float* __restrict__ denom,
                           const float* __restrict__ xl, float* __restrict__ out)
{
    long long tid = (long long)blockIdx.x * blockDim.x + threadIdx.x;
    if (tid >= (long long)E2R * 32) return;
    int e = (int)(tid >> 5), q = (int)(tid & 31);
    int src, dst;
    if (e < E_ROAD) { src = ei[e]; dst = ei[E_ROAD + e]; }
    else            { src = dst = e - E_ROAD; }
    float alpha = elog[e] / denom[dst];
    float4 xv = ((const float4*)(xl + (size_t)src * H))[q];
    float* op = out + (size_t)dst * H + q * 4;
    atomicAdd(op + 0, alpha * xv.x);
    atomicAdd(op + 1, alpha * xv.y);
    atomicAdd(op + 2, alpha * xv.z);
    atomicAdd(op + 3, alpha * xv.w);
}

__global__ void k_bias_act(float* __restrict__ y, const float* __restrict__ bias,
                           long long n4, int relu)
{
    long long tid = (long long)blockIdx.x * blockDim.x + threadIdx.x;
    if (tid >= n4) return;
    int jb = (int)(tid & 31) * 4;                // column group (H/4 == 32)
    float4 v = ((const float4*)y)[tid];
    v.x += bias[jb + 0]; v.y += bias[jb + 1];
    v.z += bias[jb + 2]; v.w += bias[jb + 3];
    if (relu) {
        v.x = fmaxf(v.x, 0.f); v.y = fmaxf(v.y, 0.f);
        v.z = fmaxf(v.z, 0.f); v.w = fmaxf(v.w, 0.f);
    }
    ((float4*)y)[tid] = v;
}

// ---------------------------------------------------------------------------
// GCN support kernels
// ---------------------------------------------------------------------------
__global__ void k_gcn_deg(const int* __restrict__ ei, const float* __restrict__ w,
                          float* __restrict__ deg)
{
    int e = blockIdx.x * blockDim.x + threadIdx.x;
    if (e >= E_ZONE) return;
    atomicAdd(&deg[ei[E_ZONE + e]], w[e]);
}

__global__ void k_gcn_dinv(float* __restrict__ deg)
{
    int i = blockIdx.x * blockDim.x + threadIdx.x;
    if (i >= N_ZONE) return;
    float d = deg[i] + 1.0f;                     // self-loop weight 1
    deg[i] = (d > 0.f) ? rsqrtf(d) : 0.f;
}

__global__ void k_gcn_aggr(const int* __restrict__ ei, const float* __restrict__ w,
                           const float* __restrict__ dinv,
                           const float* __restrict__ h, float* __restrict__ out)
{
    long long tid = (long long)blockIdx.x * blockDim.x + threadIdx.x;
    if (tid >= (long long)E2Z * 32) return;
    int e = (int)(tid >> 5), q = (int)(tid & 31);
    int src, dst; float ww;
    if (e < E_ZONE) { src = ei[e]; dst = ei[E_ZONE + e]; ww = w[e]; }
    else            { src = dst = e - E_ZONE; ww = 1.0f; }
    float norm = dinv[src] * ww * dinv[dst];
    float4 hv = ((const float4*)(h + (size_t)src * H))[q];
    float* op = out + (size_t)dst * H + q * 4;
    atomicAdd(op + 0, norm * hv.x);
    atomicAdd(op + 1, norm * hv.y);
    atomicAdd(op + 2, norm * hv.z);
    atomicAdd(op + 3, norm * hv.w);
}

// ---------------------------------------------------------------------------
// Launch
// ---------------------------------------------------------------------------
extern "C" void kernel_launch(void* const* d_in, const int* in_sizes, int n_in,
                              void* d_out, int out_size, void* d_ws, size_t ws_size,
                              hipStream_t stream)
{
    (void)in_sizes; (void)n_in; (void)out_size; (void)ws_size;

    // ---- inputs (setup_inputs() dict order) ----
    const float* id_emb   = (const float*)d_in[0];
    const float* lenW     = (const float*)d_in[1];
    const float* lenb     = (const float*)d_in[2];
    const float* type_emb = (const float*)d_in[3];
    const float* lonW     = (const float*)d_in[4];
    const float* lonb     = (const float*)d_in[5];
    const float* latW     = (const float*)d_in[6];
    const float* latb     = (const float*)d_in[7];
    const float* alen     = (const float*)d_in[8];
    const int*   atype    = (const int*)d_in[9];
    const float* alon     = (const float*)d_in[10];
    const float* alat     = (const float*)d_in[11];
    const int*   road_ei  = (const int*)d_in[12];
    const float* iattr    = (const float*)d_in[13];
    const float* g1_Wl = (const float*)d_in[14]; const float* g1_bl = (const float*)d_in[15];
    const float* g1_Wr = (const float*)d_in[16]; const float* g1_br = (const float*)d_in[17];
    const float* g1_We = (const float*)d_in[18]; const float* g1_at = (const float*)d_in[19];
    const float* g1_bs = (const float*)d_in[20];
    const float* g2_Wl = (const float*)d_in[21]; const float* g2_bl = (const float*)d_in[22];
    const float* g2_Wr = (const float*)d_in[23]; const float* g2_br = (const float*)d_in[24];
    const float* g2_We = (const float*)d_in[25]; const float* g2_at = (const float*)d_in[26];
    const float* g2_bs = (const float*)d_in[27];
    const float* zone_emb = (const float*)d_in[28];
    const int*   zone_ei  = (const int*)d_in[29];
    const float* zone_w   = (const float*)d_in[30];
    const float* c1_W = (const float*)d_in[31]; const float* c1_b = (const float*)d_in[32];
    const float* c2_W = (const float*)d_in[33]; const float* c2_b = (const float*)d_in[34];

    float* road_out = (float*)d_out;
    float* zone_out = road_out + (size_t)N_ROAD * H;

    // ---- workspace carve-up (256B aligned slots) ----
    char* wsp = (char*)d_ws;
    size_t off = 0;
    auto take = [&](size_t bytes) -> char* {
        char* p = wsp + off;
        off += (bytes + 255) & ~(size_t)255;
        return p;
    };
    _Float16*     XH     = (_Float16*)take((size_t)N_ROAD * H * 2);
    float*        XL     = (float*)take((size_t)N_ROAD * H * 4);
    float*        XR     = (float*)take((size_t)N_ROAD * H * 4);
    float*        OUT1   = (float*)take((size_t)N_ROAD * H * 4);
    _Float16*     WT     = (_Float16*)take((size_t)H * H * 2);
    float*        MEANEA = (float*)take((size_t)N_ROAD * 2 * 4);
    float*        CNT    = (float*)take((size_t)N_ROAD * 4);
    unsigned int* MAXKEY = (unsigned int*)take((size_t)N_ROAD * 4);
    float*        DENOM  = (float*)take((size_t)N_ROAD * 4);
    float*        ELOG   = (float*)take((size_t)E2R * 4);
    float*        DINV   = (float*)take((size_t)N_ZONE * 4);

    const long long NRH = (long long)N_ROAD * H;
    const long long NZH = (long long)N_ZONE * H;
    const int roadTiles = N_ROAD / 16, zoneTiles = N_ZONE / 16;
    const int gemmGridR = roadTiles < 2048 ? roadTiles : 2048;
    const int gemmGridZ = zoneTiles < 2048 ? zoneTiles : 2048;

    // ---- init (inside capture => redone every replay) ----
    hipMemsetAsync(d_out, 0, (size_t)(N_ROAD + N_ZONE) * H * 4, stream);
    hipMemsetAsync(CNT, 0, (size_t)N_ROAD * 4, stream);
    hipMemsetAsync(MEANEA, 0, (size_t)N_ROAD * 2 * 4, stream);
    hipMemsetAsync(OUT1, 0, (size_t)N_ROAD * H * 4, stream);

    // ---- road features + mean edge attr ----
    k_build_feats<<<cdiv(NRH, 256), 256, 0, stream>>>(id_emb, lenW, lenb, type_emb,
                                                      lonW, lonb, latW, latb,
                                                      alen, atype, alon, alat, XH);
    k_ea_accum<<<cdiv(E_ROAD, 256), 256, 0, stream>>>(road_ei, iattr, MEANEA, CNT);
    k_ea_div<<<cdiv(N_ROAD, 256), 256, 0, stream>>>(MEANEA, CNT);

    // ---- GAT layer (shared sequence) ----
    auto gat_layer = [&](const float* Wl, const float* bl, const float* Wr, const float* br,
                         const float* We, const float* att, const float* bias,
                         float* outbuf, int relu) {
        k_wt<<<64, 256, 0, stream>>>(Wl, WT);
        k_gemm128<<<gemmGridR, 256, 0, stream>>>(XH, WT, bl, XL, roadTiles);
        k_wt<<<64, 256, 0, stream>>>(Wr, WT);
        k_gemm128<<<gemmGridR, 256, 0, stream>>>(XH, WT, br, XR, roadTiles);
        hipMemsetAsync(MAXKEY, 0, (size_t)N_ROAD * 4, stream);
        hipMemsetAsync(DENOM, 0, (size_t)N_ROAD * 4, stream);
        k_gat_logits<<<cdiv(E2R, 8), 256, 0, stream>>>(XL, XR, road_ei, iattr, MEANEA,
                                                       We, att, ELOG, MAXKEY);
        k_gat_exp<<<cdiv(E2R, 256), 256, 0, stream>>>(road_ei, ELOG, MAXKEY, DENOM);
        k_gat_aggr<<<cdiv((long long)E2R * 32, 256), 256, 0, stream>>>(road_ei, ELOG, DENOM,
                                                                       XL, outbuf);
        k_bias_act<<<cdiv(NRH / 4, 256), 256, 0, stream>>>(outbuf, bias, NRH / 4, relu);
    };

    // layer 1 -> OUT1 (relu), then re-stage f16
    gat_layer(g1_Wl, g1_bl, g1_Wr, g1_br, g1_We, g1_at, g1_bs, OUT1, 1);
    k_f2h4<<<cdiv(NRH / 4, 256), 256, 0, stream>>>(OUT1, XH, NRH / 4);
    // layer 2 -> road_out (no relu)
    gat_layer(g2_Wl, g2_bl, g2_Wr, g2_br, g2_We, g2_at, g2_bs, road_out, 0);

    // ---- zone GCN ----
    k_f2h4<<<cdiv(NZH / 4, 256), 256, 0, stream>>>(zone_emb, XH, NZH / 4);
    hipMemsetAsync(DINV, 0, (size_t)N_ZONE * 4, stream);
    k_gcn_deg<<<cdiv(E_ZONE, 256), 256, 0, stream>>>(zone_ei, zone_w, DINV);
    k_gcn_dinv<<<cdiv(N_ZONE, 256), 256, 0, stream>>>(DINV);

    k_wt<<<64, 256, 0, stream>>>(c1_W, WT);
    k_gemm128<<<gemmGridZ, 256, 0, stream>>>(XH, WT, nullptr, XL, zoneTiles);
    hipMemsetAsync(OUT1, 0, (size_t)N_ZONE * H * 4, stream);
    k_gcn_aggr<<<cdiv((long long)E2Z * 32, 256), 256, 0, stream>>>(zone_ei, zone_w, DINV,
                                                                   XL, OUT1);
    k_bias_act<<<cdiv(NZH / 4, 256), 256, 0, stream>>>(OUT1, c1_b, NZH / 4, 1);

    k_f2h4<<<cdiv(NZH / 4, 256), 256, 0, stream>>>(OUT1, XH, NZH / 4);
    k_wt<<<64, 256, 0, stream>>>(c2_W, WT);
    k_gemm128<<<gemmGridZ, 256, 0, stream>>>(XH, WT, nullptr, XL, zoneTiles);
    k_gcn_aggr<<<cdiv((long long)E2Z * 32, 256), 256, 0, stream>>>(zone_ei, zone_w, DINV,
                                                                   XL, zone_out);
    k_bias_act<<<cdiv(NZH / 4, 256), 256, 0, stream>>>(zone_out, c2_b, NZH / 4, 0);
}
